// GPT_Network_19361712571414
// MI455X (gfx1250) — matrix operand order repaired and checked by reading, not compiled, not run on hardware
//
#include <hip/hip_runtime.h>
#include <hip/hip_bf16.h>
#include <stdint.h>

typedef __attribute__((ext_vector_type(16))) _Float16 v16h;
typedef __attribute__((ext_vector_type(8)))  float    v8f;
typedef __attribute__((ext_vector_type(4)))  float    f32x4;

#define TB 256

namespace {
constexpr int NHc = 16, Hc = 1024, FFc = 4096, Vc = 32000, Sc = 1024, Ec = 128;
constexpr int Bc = 4, HEADc = 64, LLc = 11, BSc = Bc * Sc;  // 4096 rows
constexpr int F_OUT16 = 1, F_RELU = 2, F_NT = 4;
constexpr int LPITCH = 72;                 // f16 pitch per LDS row (144B, 16B aligned)
constexpr uint32_t LBUF = 128 * LPITCH * 2; // bytes per LDS buffer (A or B), 18432
}

// ---------------------------------------------------------------------------
// Generic WMMA GEMM:  C[m,n] = scale * sum_k A[m,k] * Bt[n,k]  (+bias, relu)
// A: f16 row-major [M,K] stride lda ; Bt: f16 row-major [N,K] stride ldb
// Batched via blockIdx.z with two-level stride decomposition (z/zdiv, z%zdiv).
// Tile: 128x128 per WG, BK=64, 8 wave32s each computing 32x64.
// Double-buffered LDS fed by GLOBAL_LOAD_ASYNC_TO_LDS_B128 (ASYNCcnt path):
// per iteration: s_wait_asynccnt 0 -> barrier -> issue next tile DMA -> 16 WMMAs.
// ---------------------------------------------------------------------------
__global__ __launch_bounds__(TB) void gemm_wmma(
    const _Float16* __restrict__ A, long lda, long aS1, long aS2,
    const _Float16* __restrict__ Bm, long ldb, long bS1, long bS2,
    void* __restrict__ Cv, long ldc, long cS1, long cS2,
    int M, int N, int K, int zdiv,
    const float* __restrict__ bias, float scale, int flags)
{
    __shared__ _Float16 sA[2][128 * LPITCH];
    __shared__ _Float16 sB[2][128 * LPITCH];

    const int z = blockIdx.z;
    A  += (long)(z / zdiv) * aS1 + (long)(z % zdiv) * aS2;
    Bm += (long)(z / zdiv) * bS1 + (long)(z % zdiv) * bS2;
    const long cOff = (long)(z / zdiv) * cS1 + (long)(z % zdiv) * cS2;

    const int tid   = threadIdx.x;
    const int mBase = blockIdx.y * 128;
    const int nBase = blockIdx.x * 128;
    const int wid   = tid >> 5;
    const int lane  = tid & 31;
    const int wm    = (wid & 3) * 32;   // wave M offset in tile
    const int wn    = (wid >> 2) * 64;  // wave N offset in tile
    const int lm    = lane & 15;        // lane row/col within fragment
    const int lg    = lane >> 4;        // lane half-group

    // --- async staging addresses (4 x 16B chunks per thread per matrix) ---
    const int srow = tid >> 3;          // 0..31
    const int scol = (tid & 7) * 8;     // element column within BK=64
    uint32_t lA[4], lB[4];              // LDS byte offsets (buffer 0)
    int32_t  gA[4], gB[4];              // global byte offsets from tile base
    #pragma unroll
    for (int i = 0; i < 4; ++i) {
        int row = i * 32 + srow;
        int ga = mBase + row; if (ga > M - 1) ga = M - 1;
        int gb = nBase + row; if (gb > N - 1) gb = N - 1;
        lA[i] = (uint32_t)(uintptr_t)&sA[0][row * LPITCH + scol];
        lB[i] = (uint32_t)(uintptr_t)&sB[0][row * LPITCH + scol];
        gA[i] = (int32_t)(((long)ga * lda + scol) * 2);
        gB[i] = (int32_t)(((long)gb * ldb + scol) * 2);
    }
    const _Float16* pa = A;   // SADDR, bumped 64 elements (128B) per K-tile
    const _Float16* pb = Bm;

    auto issue = [&](int buf, const _Float16* qa, const _Float16* qb) {
        const uint32_t bo = buf ? LBUF : 0u;
        #pragma unroll
        for (int i = 0; i < 4; ++i) {
            asm volatile("global_load_async_to_lds_b128 %0, %1, %2"
                         :: "v"(lA[i] + bo), "v"(gA[i]), "s"(qa) : "memory");
            asm volatile("global_load_async_to_lds_b128 %0, %1, %2"
                         :: "v"(lB[i] + bo), "v"(gB[i]), "s"(qb) : "memory");
        }
    };

    v8f acc[2][4] = {};
    const int nk = K >> 6;

    issue(0, pa, pb);                       // prologue: tile 0 -> buffer 0
    for (int kt = 0; kt < nk; ++kt) {
        asm volatile("s_wait_asynccnt 0" ::: "memory");  // my tile-kt DMA done
        __syncthreads();                    // all waves: tile kt visible, buf (kt+1)&1 free
        if (kt + 1 < nk) {
            pa += 64; pb += 64;
            issue((kt + 1) & 1, pa, pb);    // overlap tile kt+1 DMA with compute
        }
        const _Float16* cA = sA[kt & 1];
        const _Float16* cB = sB[kt & 1];

        #pragma unroll
        for (int ks = 0; ks < 2; ++ks) {    // two K=32 WMMA steps per BK=64 tile
            v16h af[2], bf[4];
            #pragma unroll
            for (int i = 0; i < 2; ++i) {
                // A frag: lane lm = row M, VGPR0-3: K = g*8.., VGPR4-7: K = 16+g*8..
                const _Float16* p = &cA[(wm + i * 16 + lm) * LPITCH + ks * 32 + lg * 8];
                ((f32x4*)&af[i])[0] = *(const f32x4*)(p);
                ((f32x4*)&af[i])[1] = *(const f32x4*)(p + 16);
            }
            #pragma unroll
            for (int j = 0; j < 4; ++j) {
                // B frag: lane lm = column N, VGPR0-3: K = g*8.., VGPR4-7: K = 16+g*8..
                const _Float16* p = &cB[(wn + j * 16 + lm) * LPITCH + ks * 32 + lg * 8];
                ((f32x4*)&bf[j])[0] = *(const f32x4*)(p);
                ((f32x4*)&bf[j])[1] = *(const f32x4*)(p + 16);
            }
            #pragma unroll
            for (int i = 0; i < 2; ++i)
                #pragma unroll
                for (int j = 0; j < 4; ++j)
                    acc[i][j] = __builtin_amdgcn_wmma_f32_16x16x32_f16(
                        false, af[i], false, bf[j], (short)0, acc[i][j], false, false);
        }
    }

    const bool outF16 = (flags & F_OUT16) != 0;
    const bool doRelu = (flags & F_RELU) != 0;
    const bool doNT   = (flags & F_NT) != 0;
    #pragma unroll
    for (int j = 0; j < 4; ++j) {
        int n = nBase + wn + j * 16 + lm;
        if (n >= N) continue;
        float bv = bias ? bias[n] : 0.0f;
        #pragma unroll
        for (int i = 0; i < 2; ++i) {
            #pragma unroll
            for (int r = 0; r < 8; ++r) {
                int m = mBase + wm + i * 16 + lg * 8 + r;  // C layout: VGPR r -> M = 8*g + r
                if (m >= M) continue;
                float v = acc[i][j][r] * scale + bv;
                if (doRelu) v = v > 0.0f ? v : 0.0f;
                long idx = cOff + (long)m * ldc + n;
                if (outF16)      ((_Float16*)Cv)[idx] = (_Float16)v;
                else if (doNT)   __builtin_nontemporal_store(v, &((float*)Cv)[idx]);
                else             ((float*)Cv)[idx] = v;
            }
        }
    }
}

// ---------------------------------------------------------------------------
// Weight convert + transpose: dst f16 [N,K] <- src f32 [K,N]
// ---------------------------------------------------------------------------
__global__ void wtrans(const float* __restrict__ src, _Float16* __restrict__ dst,
                       int K, int N, long total)
{
    long t = (long)blockIdx.x * TB + threadIdx.x;
    if (t >= total) return;
    long n = t / K, k = t - n * K;
    dst[t] = (_Float16)src[k * (long)N + n];
}

// Embedding gather -> f16 [BS, E]
__global__ void embed_gather(const int* __restrict__ xi, const float* __restrict__ emb,
                             _Float16* __restrict__ out)
{
    long t = (long)blockIdx.x * TB + threadIdx.x;
    int r = (int)(t >> 7);
    int c = (int)(t & (Ec - 1));
    out[t] = (_Float16)emb[(long)xi[r] * Ec + c];
}

// layer_in f16 = x_emb + pos[11]  (pos pre-offset to layer 11)
__global__ void lin16_kernel(const float* __restrict__ xemb, const float* __restrict__ pos,
                             _Float16* __restrict__ out)
{
    long t = (long)blockIdx.x * TB + threadIdx.x;
    int  c = (int)(t & (Hc - 1));
    long r = t >> 10;
    int  s = (int)(r & (Sc - 1));
    out[t] = (_Float16)(xemb[t] + pos[(long)s * Hc + c]);
}

// V [BS,H] f16 -> Vt [B,NH,HEAD,S] f16
__global__ void vtrans_kernel(const _Float16* __restrict__ v, _Float16* __restrict__ vt)
{
    long t = (long)blockIdx.x * TB + threadIdx.x;
    int  j = (int)(t & (Sc - 1));
    long rest = t >> 10;
    int  d = (int)(rest & (HEADc - 1));
    int  h = (int)(rest >> 6);
    int  b = h >> 4, nh = h & 15;
    vt[t] = v[((long)(b * Sc + j)) * Hc + nh * HEADc + d];
}

// In-place causal softmax over f16 score rows [B*NH, S, S]
__global__ __launch_bounds__(TB) void softmax_causal(_Float16* __restrict__ sc)
{
    const int  i = blockIdx.x;
    const long h = blockIdx.y;
    _Float16* row = sc + (h * Sc + (long)i) * Sc;
    __shared__ float red[TB];
    const int tid = threadIdx.x;

    float vals[4];
    float vmax = -3.0e38f;
    #pragma unroll
    for (int t = 0; t < 4; ++t) {
        int j = tid + t * TB;
        float v = (j <= i) ? (float)row[j] : -3.0e38f;
        vals[t] = v;
        vmax = fmaxf(vmax, v);
    }
    red[tid] = vmax; __syncthreads();
    for (int s2 = TB / 2; s2 > 0; s2 >>= 1) {
        if (tid < s2) red[tid] = fmaxf(red[tid], red[tid + s2]);
        __syncthreads();
    }
    vmax = red[0]; __syncthreads();

    float sum = 0.f;
    #pragma unroll
    for (int t = 0; t < 4; ++t) {
        int j = tid + t * TB;
        float e = (j <= i) ? __expf(vals[t] - vmax) : 0.f;
        vals[t] = e; sum += e;
    }
    red[tid] = sum; __syncthreads();
    for (int s2 = TB / 2; s2 > 0; s2 >>= 1) {
        if (tid < s2) red[tid] += red[tid + s2];
        __syncthreads();
    }
    float inv = 1.0f / red[0];
    #pragma unroll
    for (int t = 0; t < 4; ++t) {
        int j = tid + t * TB;
        row[j] = (_Float16)(vals[t] * inv);
    }
}

// x_self_norm = (xemb + pos) + LN(x_multi; bias1, scale1); write f32 + f16
__global__ __launch_bounds__(TB) void ln1_kernel(
    const float* __restrict__ xmulti, const float* __restrict__ xemb,
    const float* __restrict__ pos, const float* __restrict__ bias,
    const float* __restrict__ scl, float* __restrict__ xself,
    _Float16* __restrict__ xself16)
{
    const long r = blockIdx.x;
    const int  s = (int)(r & (Sc - 1));
    const float* xm = xmulti + r * Hc;
    __shared__ float red[TB];
    const int tid = threadIdx.x;

    float v4[4], sum = 0.f;
    #pragma unroll
    for (int t = 0; t < 4; ++t) { int j = tid + t * TB; v4[t] = xm[j]; sum += v4[t]; }
    red[tid] = sum; __syncthreads();
    for (int s2 = TB / 2; s2 > 0; s2 >>= 1) { if (tid < s2) red[tid] += red[tid + s2]; __syncthreads(); }
    float mean = red[0] * (1.0f / Hc); __syncthreads();

    float vs = 0.f;
    #pragma unroll
    for (int t = 0; t < 4; ++t) { float d = v4[t] - mean; vs += d * d; }
    red[tid] = vs; __syncthreads();
    for (int s2 = TB / 2; s2 > 0; s2 >>= 1) { if (tid < s2) red[tid] += red[tid + s2]; __syncthreads(); }
    float rstd = rsqrtf(red[0] * (1.0f / Hc) + 1e-6f);

    #pragma unroll
    for (int t = 0; t < 4; ++t) {
        int j = tid + t * TB;
        float base = xemb[r * Hc + j] + pos[(long)s * Hc + j];
        float o = base + (v4[t] - mean) * rstd * scl[j] + bias[j];
        xself[r * Hc + j]   = o;
        xself16[r * Hc + j] = (_Float16)o;
    }
}

// Fused: t = xself + LN(ffw2 + bff2; bias2, scale2); out = xemb + LN(t; obias, oscale) -> f16
__global__ __launch_bounds__(TB) void ln2_kernel(
    const float* __restrict__ ffw2, const float* __restrict__ bff2,
    const float* __restrict__ bias2, const float* __restrict__ scl2,
    const float* __restrict__ xself, const float* __restrict__ xemb,
    const float* __restrict__ obias, const float* __restrict__ oscl,
    _Float16* __restrict__ dec16)
{
    const long r = blockIdx.x;
    __shared__ float red[TB];
    const int tid = threadIdx.x;

    float v4[4], sum = 0.f;
    #pragma unroll
    for (int t = 0; t < 4; ++t) { int j = tid + t * TB; v4[t] = ffw2[r * Hc + j] + bff2[j]; sum += v4[t]; }
    red[tid] = sum; __syncthreads();
    for (int s2 = TB / 2; s2 > 0; s2 >>= 1) { if (tid < s2) red[tid] += red[tid + s2]; __syncthreads(); }
    float mean = red[0] * (1.0f / Hc); __syncthreads();

    float vs = 0.f;
    #pragma unroll
    for (int t = 0; t < 4; ++t) { float d = v4[t] - mean; vs += d * d; }
    red[tid] = vs; __syncthreads();
    for (int s2 = TB / 2; s2 > 0; s2 >>= 1) { if (tid < s2) red[tid] += red[tid + s2]; __syncthreads(); }
    float rstd = rsqrtf(red[0] * (1.0f / Hc) + 1e-6f); __syncthreads();

    sum = 0.f;
    #pragma unroll
    for (int t = 0; t < 4; ++t) {
        int j = tid + t * TB;
        v4[t] = xself[r * Hc + j] + (v4[t] - mean) * rstd * scl2[j] + bias2[j];
        sum += v4[t];
    }
    red[tid] = sum; __syncthreads();
    for (int s2 = TB / 2; s2 > 0; s2 >>= 1) { if (tid < s2) red[tid] += red[tid + s2]; __syncthreads(); }
    float mean2 = red[0] * (1.0f / Hc); __syncthreads();

    vs = 0.f;
    #pragma unroll
    for (int t = 0; t < 4; ++t) { float d = v4[t] - mean2; vs += d * d; }
    red[tid] = vs; __syncthreads();
    for (int s2 = TB / 2; s2 > 0; s2 >>= 1) { if (tid < s2) red[tid] += red[tid + s2]; __syncthreads(); }
    float rstd2 = rsqrtf(red[0] * (1.0f / Hc) + 1e-6f);

    #pragma unroll
    for (int t = 0; t < 4; ++t) {
        int j = tid + t * TB;
        float o = xemb[r * Hc + j] + (v4[t] - mean2) * rstd2 * oscl[j] + obias[j];
        dec16[r * Hc + j] = (_Float16)o;
    }
}

// ---------------------------------------------------------------------------
extern "C" void kernel_launch(void* const* d_in, const int* in_sizes, int n_in,
                              void* d_out, int out_size, void* d_ws, size_t ws_size,
                              hipStream_t stream)
{
    (void)in_sizes; (void)n_in; (void)out_size; (void)ws_size;

    const int*   xi    = (const int*)  d_in[0];
    const float* emb   = (const float*)d_in[1];
    const float* Wdec  = (const float*)d_in[2];
    const float* Wproj = (const float*)d_in[3];
    const float* pos   = (const float*)d_in[4]  + (long)LLc * Sc * Hc;
    const float* Wq    = (const float*)d_in[5]  + (long)LLc * Hc * Hc;
    const float* Wk    = (const float*)d_in[6]  + (long)LLc * Hc * Hc;
    const float* Wv    = (const float*)d_in[7]  + (long)LLc * Hc * Hc;
    const float* Wc    = (const float*)d_in[8]  + (long)LLc * Hc * Hc;
    const float* Wff1  = (const float*)d_in[9]  + (long)LLc * Hc * FFc;
    const float* Wff2  = (const float*)d_in[10] + (long)LLc * FFc * Hc;
    const float* bff1  = (const float*)d_in[11] + (long)LLc * FFc;
    const float* bff2  = (const float*)d_in[12] + (long)LLc * Hc;
    const float* obias = (const float*)d_in[13];
    const float* oscl  = (const float*)d_in[14];
    const float* bias1 = (const float*)d_in[15] + (long)LLc * Hc;
    const float* bias2 = (const float*)d_in[16] + (long)LLc * Hc;
    const float* scl1  = (const float*)d_in[17] + (long)LLc * Hc;
    const float* scl2  = (const float*)d_in[18] + (long)LLc * Hc;

    // Workspace bump allocator (256B aligned)
    char* wsb = (char*)d_ws;
    size_t off = 0;
    auto alloc = [&](size_t bytes) -> char* {
        off = (off + 255) & ~(size_t)255;
        char* p = wsb + off;
        off += bytes;
        return p;
    };
    _Float16* WQt    = (_Float16*)alloc((size_t)Hc * Hc * 2);
    _Float16* WKt    = (_Float16*)alloc((size_t)Hc * Hc * 2);
    _Float16* WVt    = (_Float16*)alloc((size_t)Hc * Hc * 2);
    _Float16* WCt    = (_Float16*)alloc((size_t)Hc * Hc * 2);
    _Float16* WFF1t  = (_Float16*)alloc((size_t)FFc * Hc * 2);
    _Float16* WFF2t  = (_Float16*)alloc((size_t)Hc * FFc * 2);
    _Float16* WDECt  = (_Float16*)alloc((size_t)Hc * Ec * 2);
    _Float16* WPROJt = (_Float16*)alloc((size_t)Vc * Hc * 2);
    _Float16* EMB16  = (_Float16*)alloc((size_t)BSc * Ec * 2);
    float*    XEMB   = (float*)   alloc((size_t)BSc * Hc * 4);
    _Float16* LIN16  = (_Float16*)alloc((size_t)BSc * Hc * 2);
    _Float16* Q16    = (_Float16*)alloc((size_t)BSc * Hc * 2);
    _Float16* K16    = (_Float16*)alloc((size_t)BSc * Hc * 2);
    _Float16* V16    = (_Float16*)alloc((size_t)BSc * Hc * 2);
    _Float16* VT16   = (_Float16*)alloc((size_t)BSc * Hc * 2);
    _Float16* SC16   = (_Float16*)alloc((size_t)Bc * NHc * Sc * Sc * 2);  // 128 MB
    _Float16* CONC16 = (_Float16*)alloc((size_t)BSc * Hc * 2);
    float*    XMULTI = (float*)   alloc((size_t)BSc * Hc * 4);
    float*    XSELF  = (float*)   alloc((size_t)BSc * Hc * 4);
    _Float16* XSELF16= (_Float16*)alloc((size_t)BSc * Hc * 2);
    // Stream-ordered reuse (attention scores are consumed before FF starts):
    _Float16* FFW1   = SC16;                                              // 32 MB
    float*    FFW2   = (float*)((char*)SC16 + (size_t)32 * 1024 * 1024);  // 16 MB
    _Float16* DEC16  = Q16;                                               // Q dead after scores

    auto tr = [&](const float* s, _Float16* d, int K, int N) {
        long tot = (long)K * N;
        wtrans<<<(unsigned)((tot + TB - 1) / TB), TB, 0, stream>>>(s, d, K, N, tot);
    };
    auto gemm = [&](const _Float16* A, long lda, long aS1, long aS2,
                    const _Float16* Bp, long ldb, long bS1, long bS2,
                    void* C, long ldc, long cS1, long cS2,
                    int M, int N, int K, int Z, int zdiv,
                    const float* bias, float scale, int flags) {
        dim3 g((N + 127) / 128, (M + 127) / 128, Z);
        gemm_wmma<<<g, TB, 0, stream>>>(A, lda, aS1, aS2, Bp, ldb, bS1, bS2,
                                        C, ldc, cS1, cS2, M, N, K, zdiv, bias, scale, flags);
    };

    // 1. Weight prep (convert + transpose to [N,K] f16)
    tr(Wq, WQt, Hc, Hc);     tr(Wk, WKt, Hc, Hc);
    tr(Wv, WVt, Hc, Hc);     tr(Wc, WCt, Hc, Hc);
    tr(Wff1, WFF1t, Hc, FFc); tr(Wff2, WFF2t, FFc, Hc);
    tr(Wdec, WDECt, Ec, Hc);  tr(Wproj, WPROJt, Hc, Vc);

    // 2. Embedding: gather + GEMM -> x_dec_embed (f32), layer_in f16
    embed_gather<<<(BSc * Ec) / TB, TB, 0, stream>>>(xi, emb, EMB16);
    gemm(EMB16, Ec, 0, 0, WDECt, Ec, 0, 0, XEMB, Hc, 0, 0,
         BSc, Hc, Ec, 1, 1, nullptr, 1.0f, 0);
    lin16_kernel<<<(BSc * Hc) / TB, TB, 0, stream>>>(XEMB, pos, LIN16);

    // 3. QKV (Q pre-scaled by 1/sqrt(HEAD))
    gemm(LIN16, Hc, 0, 0, WQt, Hc, 0, 0, Q16, Hc, 0, 0, BSc, Hc, Hc, 1, 1, nullptr, 0.125f, F_OUT16);
    gemm(LIN16, Hc, 0, 0, WKt, Hc, 0, 0, K16, Hc, 0, 0, BSc, Hc, Hc, 1, 1, nullptr, 1.0f, F_OUT16);
    gemm(LIN16, Hc, 0, 0, WVt, Hc, 0, 0, V16, Hc, 0, 0, BSc, Hc, Hc, 1, 1, nullptr, 1.0f, F_OUT16);
    vtrans_kernel<<<(BSc * Hc) / TB, TB, 0, stream>>>(V16, VT16);

    // 4. Attention: scores = q k^T (batched over 64 heads, z=(b,nh))
    gemm(Q16, Hc, (long)Sc * Hc, HEADc,
         K16, Hc, (long)Sc * Hc, HEADc,
         SC16, Sc, (long)NHc * Sc * Sc, (long)Sc * Sc,
         Sc, Sc, HEADc, Bc * NHc, NHc, nullptr, 1.0f, F_OUT16);
    softmax_causal<<<dim3(Sc, Bc * NHc), TB, 0, stream>>>(SC16);
    // context = alphas @ v  -> combined heads directly into [BS,H]
    gemm(SC16, Sc, (long)NHc * Sc * Sc, (long)Sc * Sc,
         VT16, Sc, (long)NHc * HEADc * Sc, (long)HEADc * Sc,
         CONC16, Hc, (long)Sc * Hc, HEADc,
         Sc, HEADc, Sc, Bc * NHc, NHc, nullptr, 1.0f, F_OUT16);

    // 5. Output projection of attention + LN residual
    gemm(CONC16, Hc, 0, 0, WCt, Hc, 0, 0, XMULTI, Hc, 0, 0,
         BSc, Hc, Hc, 1, 1, nullptr, 1.0f, 0);
    ln1_kernel<<<BSc, TB, 0, stream>>>(XMULTI, XEMB, pos, bias1, scl1, XSELF, XSELF16);

    // 6. FFN (bias+relu fused into GEMM epilogue)
    gemm(XSELF16, Hc, 0, 0, WFF1t, Hc, 0, 0, FFW1, FFc, 0, 0,
         BSc, FFc, Hc, 1, 1, bff1, 1.0f, F_OUT16 | F_RELU);
    gemm(FFW1, FFc, 0, 0, WFF2t, FFc, 0, 0, FFW2, Hc, 0, 0,
         BSc, Hc, FFc, 1, 1, nullptr, 1.0f, 0);

    // 7. Fused double layer-norm -> dec_outputs f16
    ln2_kernel<<<BSc, TB, 0, stream>>>(FFW2, bff2, bias2, scl2, XSELF, XEMB,
                                       obias, oscl, DEC16);

    // 8. Vocab projection -> f32 logits, non-temporal stores (write-once stream)
    gemm(DEC16, Hc, 0, 0, WPROJt, Hc, 0, 0, (float*)d_out, Vc, 0, 0,
         BSc, Vc, Hc, 1, 1, nullptr, 1.0f, F_NT);
}
